// TransformerBlock_91225105367666
// MI455X (gfx1250) — compile-verified, matrix operand order checked by reading, not run on hardware
//
#include <hip/hip_runtime.h>
#include <math.h>

// ---------------------------------------------------------------------------
// Problem constants (from reference): B=8, N=2048, D_POINTS=64, D_MODEL=512, k=16
// ---------------------------------------------------------------------------
#define BDIM 8
#define NPTS 2048
#define DPTS 64
#define DMOD 512
#define KNN  16
constexpr int NM = BDIM * NPTS; // 16384 total points

typedef unsigned short u16;
typedef __bf16 bf16_t;
typedef bf16_t v16bf __attribute__((ext_vector_type(16)));
typedef float  v8f   __attribute__((ext_vector_type(8)));

union BF16Frag { v16bf v; u16 u[16]; uint4 q[2]; };

__device__ __forceinline__ u16 f2bf(float f) {
  union { float f; unsigned u; } x; x.f = f;
  unsigned r = x.u + 0x7FFFu + ((x.u >> 16) & 1u); // round-to-nearest-even
  return (u16)(r >> 16);
}

// K-offset of VGPR i within a 16-bit 16x32 A fragment (ISA 7.12.2):
// lanes 0-15 hold K={0..7,16..23}, lanes 16-31 hold K={8..15,24..31}
__device__ __forceinline__ int frag_k(int i, int hf) {
  return ((i < 4) ? (2 * i) : (16 + 2 * (i - 4))) + hf * 8;
}

// A fragment from row-major 16xK tile of bf16 (also valid as B fragment when
// the logical B is stored transposed/row-major, e.g. S = Q * K^T).
// Per lane this is two contiguous 16B chunks -> vectorizes to b128 loads.
__device__ __forceinline__ v16bf load_frag_bf16(const u16* p, int ld, int lane) {
  const int hf = lane >> 4, r = lane & 15;
  BF16Frag f;
#pragma unroll
  for (int i = 0; i < 8; ++i) {
    const int kk = frag_k(i, hf);
    f.u[2 * i]     = p[r * ld + kk];
    f.u[2 * i + 1] = p[r * ld + kk + 1];
  }
  return f.v;
}

__device__ __forceinline__ v16bf load_frag_f32(const float* p, int ld, int lane) {
  const int hf = lane >> 4, r = lane & 15;
  BF16Frag f;
#pragma unroll
  for (int i = 0; i < 8; ++i) {
    const int kk = frag_k(i, hf);
    f.u[2 * i]     = f2bf(p[r * ld + kk]);
    f.u[2 * i + 1] = f2bf(p[r * ld + kk + 1]);
  }
  return f.v;
}

// B fragment from pre-packed fragment-major layout: per (ktile,ctile) tile of
// 32(K)x16(N), 32 lanes x 16 bf16 contiguous -> exactly two b128 loads.
__device__ __forceinline__ v16bf load_fragB_packed(const u16* tile, int lane) {
  BF16Frag f;
  const uint4* q = (const uint4*)(tile + lane * 16);
  f.q[0] = q[0];
  f.q[1] = q[1];
  return f.v;
}

__device__ __forceinline__ v8f wmma_bf16(v16bf a, v16bf b, v8f c) {
  return __builtin_amdgcn_wmma_f32_16x16x32_bf16(false, a, false, b, (short)0, c,
                                                 false, false);
}

// ---------------------------------------------------------------------------
// Pack a row-major B[K,N] operand (f32 or bf16 source) into fragment-major
// bf16 tiles: out[((kt*(N/16))+ct)*512 + lane*16 + e]  (tile = 32x16, 512 u16)
// ---------------------------------------------------------------------------
template <bool SRC_BF16>
__global__ __launch_bounds__(256) void pack_b_kernel(const void* __restrict__ src,
                                                     u16* __restrict__ out,
                                                     int K, int N) {
  const int idx = blockIdx.x * 256 + threadIdx.x;
  if (idx >= K * N) return;
  const int e    = idx & 15;
  const int lane = (idx >> 4) & 31;
  const int rest = idx >> 9;
  const int ntile = N >> 4;
  const int ct = rest % ntile;
  const int kt = rest / ntile;
  const int kk  = kt * 32 + frag_k(e >> 1, lane >> 4) + (e & 1);
  const int col = ct * 16 + (lane & 15);
  u16 v;
  if (SRC_BF16) v = ((const u16*)src)[(size_t)kk * N + col];
  else          v = f2bf(((const float*)src)[(size_t)kk * N + col]);
  out[idx] = v;
}

// ---------------------------------------------------------------------------
// K1: kNN. One wave32 per query point. Per-lane register top-16 (unrolled
// insertion), then 16 rounds of cross-lane argmin merge via shfl_xor.
// Emits normalized inverse-distance weights (weights sum to 1 -> the weighted
// average can be hoisted before the Wd2 GEMM since it is linear).
// ---------------------------------------------------------------------------
__global__ __launch_bounds__(256) void knn_kernel(const float* __restrict__ xyz,
                                                  int* __restrict__ kidx,
                                                  float* __restrict__ kw) {
  const int gw   = blockIdx.x * 8 + (threadIdx.x >> 5);
  const int lane = threadIdx.x & 31;
  const int b = gw / NPTS, n = gw % NPTS;
  const float* q = xyz + (size_t)(b * NPTS + n) * 3;
  const float qx = q[0], qy = q[1], qz = q[2];

  float best[KNN];
  int   bidx[KNN];
#pragma unroll
  for (int j = 0; j < KNN; ++j) { best[j] = 3.0e38f; bidx[j] = 0; }

  for (int m = lane; m < NPTS; m += 32) {
    const float* p = xyz + (size_t)(b * NPTS + m) * 3;
    const float dx = p[0] - qx, dy = p[1] - qy, dz = p[2] - qz;
    const float d = dx * dx + dy * dy + dz * dz;
    if (d < best[KNN - 1]) {
      best[KNN - 1] = d; bidx[KNN - 1] = m;
#pragma unroll
      for (int j = KNN - 1; j > 0; --j) {
        if (best[j] < best[j - 1]) {
          float td = best[j]; best[j] = best[j - 1]; best[j - 1] = td;
          int   ti = bidx[j]; bidx[j] = bidx[j - 1]; bidx[j - 1] = ti;
        }
      }
    }
  }

  // cross-lane merge: 16 rounds of wave argmin over each lane's head element
  int pos = 0;
  float myd = 3.0e38f; int myi = 0;
  for (int rnd = 0; rnd < KNN; ++rnd) {
    float h = 3.0e38f; int hi = 0;
#pragma unroll
    for (int j = 0; j < KNN; ++j)
      if (j == pos) { h = best[j]; hi = bidx[j]; }
    float vmin = h; int vlane = lane; int vi = hi;
#pragma unroll
    for (int off = 16; off > 0; off >>= 1) {
      float od = __shfl_xor(vmin, off, 32);
      int   ol = __shfl_xor(vlane, off, 32);
      int   oi = __shfl_xor(vi, off, 32);
      if (od < vmin || (od == vmin && ol < vlane)) { vmin = od; vlane = ol; vi = oi; }
    }
    if (lane == vlane) pos++;
    if (lane == rnd) { myd = vmin; myi = vi; }
  }

  float w = (lane < KNN) ? (1.0f / (myd + 1e-6f)) : 0.0f;
  float s = w;
#pragma unroll
  for (int off = 16; off > 0; off >>= 1) s += __shfl_xor(s, off, 32);
  if (lane < KNN) {
    kidx[(size_t)(b * NPTS + n) * KNN + lane] = myi;
    kw  [(size_t)(b * NPTS + n) * KNN + lane] = w / s;
  }
}

// ---------------------------------------------------------------------------
// K2: hoisted pos-enc hidden layer. h[q,c] = sum_k w'_k * relu(xyz_k . Wd1[:,c] + bd1[c])
// (weighted average moved before the Wd2 GEMM -> shrinks dominant GEMM 16x)
// ---------------------------------------------------------------------------
__global__ __launch_bounds__(256) void posenc_kernel(const float* __restrict__ xyz,
                                                     const int* __restrict__ kidx,
                                                     const float* __restrict__ kw,
                                                     const float* __restrict__ Wd1,
                                                     const float* __restrict__ bd1,
                                                     u16* __restrict__ h_out) {
  const int q = blockIdx.x;          // 0..NM-1
  const int b = q / NPTS;
  __shared__ float nx[KNN * 3];
  __shared__ float nw[KNN];
  if (threadIdx.x < KNN) {
    const int idx = kidx[(size_t)q * KNN + threadIdx.x];
    nw[threadIdx.x] = kw[(size_t)q * KNN + threadIdx.x];
    const float* p = xyz + (size_t)(b * NPTS + idx) * 3;
    nx[threadIdx.x * 3 + 0] = p[0];
    nx[threadIdx.x * 3 + 1] = p[1];
    nx[threadIdx.x * 3 + 2] = p[2];
  }
  __syncthreads();
  for (int c = threadIdx.x; c < DMOD; c += 256) {
    const float w0 = Wd1[c], w1 = Wd1[DMOD + c], w2 = Wd1[2 * DMOD + c], bb = bd1[c];
    float s = 0.f;
#pragma unroll
    for (int k = 0; k < KNN; ++k) {
      const float h = nx[k * 3] * w0 + nx[k * 3 + 1] * w1 + nx[k * 3 + 2] * w2 + bb;
      s += nw[k] * fmaxf(h, 0.f);
    }
    h_out[(size_t)q * DMOD + c] = f2bf(s);
  }
}

// ---------------------------------------------------------------------------
// Generic WMMA GEMM: out[M,N] = A[M,K] @ B[K,N] (+bias[N]) (+add[M,N]).
// Each wave computes a 16x64 tile (A fragment reused across 4 B tiles).
// A is bf16 (ws) or f32 (global, converted on load); B is pre-packed bf16.
// ---------------------------------------------------------------------------
template <bool A_BF16, bool OUT_BF16, bool HAS_BIAS, bool HAS_ADD>
__global__ __launch_bounds__(256) void gemm_kernel(const void* __restrict__ Ap,
                                                   const u16* __restrict__ Bp,
                                                   const float* __restrict__ bias,
                                                   const float* __restrict__ addv,
                                                   void* __restrict__ outp,
                                                   int M, int N, int K) {
  const int lane = threadIdx.x & 31;
  const int gw = blockIdx.x * 8 + (threadIdx.x >> 5);
  const int nt_count = N >> 6;
  const int mt = gw / nt_count;
  const int nt = gw - mt * nt_count;
  if (mt * 16 >= M) return; // wave-uniform
  const int row0 = mt * 16, col0 = nt * 64;
  const int ntile = N >> 4;

  v8f acc[4] = {};
  for (int kk = 0; kk < K; kk += 32) {
    v16bf a;
    if (A_BF16) a = load_frag_bf16((const u16*)Ap + (size_t)row0 * K + kk, K, lane);
    else        a = load_frag_f32((const float*)Ap + (size_t)row0 * K + kk, K, lane);
    const u16* btile = Bp + ((size_t)(kk >> 5) * ntile + (col0 >> 4)) * 512;
#pragma unroll
    for (int t = 0; t < 4; ++t) {
      v16bf bb = load_fragB_packed(btile + t * 512, lane);
      acc[t] = wmma_bf16(a, bb, acc[t]);
    }
  }
  const int hf = lane >> 4, r = lane & 15;
#pragma unroll
  for (int t = 0; t < 4; ++t) {
#pragma unroll
    for (int j = 0; j < 8; ++j) {
      const int row = row0 + j + hf * 8;
      const int col = col0 + t * 16 + r;
      float v = acc[t][j];
      if (HAS_BIAS) v += bias[col];
      if (HAS_ADD)  v += addv[(size_t)row * N + col];
      if (OUT_BF16) ((u16*)outp)[(size_t)row * N + col] = f2bf(v);
      else          ((float*)outp)[(size_t)row * N + col] = v;
    }
  }
}

// ---------------------------------------------------------------------------
// K6: flash attention. One workgroup (8 wave32) per 16 query rows.
// Q tile in LDS; loop over 256-key chunks:
//   phase1: each wave -> 16x32 slab of S = (Q)(K^T)*scale via WMMA (B^T
//           fragments use the A-fragment pattern on row-major K)
//   phase2: online softmax (row max / rescale / exp -> P bf16 in LDS)
//   phase3: each wave owns a 64-wide D slice, O += P*V in register v8f accs;
//           V comes from the pre-packed fragment-major copy (b128 loads).
// Output O/l streamed to d_out attention section (f32).
// ---------------------------------------------------------------------------
__global__ __launch_bounds__(256) void attn_kernel(const u16* __restrict__ qp,
                                                   const u16* __restrict__ kp,
                                                   const u16* __restrict__ vpack,
                                                   float* __restrict__ out) {
  __shared__ u16   Qs[16 * DMOD];   // 16 KB
  __shared__ float Ss[16 * 256];    // 16 KB
  __shared__ u16   Ps[16 * 256];    // 8 KB
  __shared__ float red[16 * 16];    // 1 KB
  __shared__ float m_i[16], l_i[16], sc_i[16];

  const int tid = threadIdx.x;
  const int lane = tid & 31;
  const int wave = tid >> 5;
  const int b = blockIdx.x >> 7;        // / (NPTS/16)
  const int mt = blockIdx.x & 127;
  const int row0 = mt * 16;
  const float scale = 0.04419417382415922f; // 1/sqrt(512)

  for (int i = tid; i < 16 * DMOD; i += 256) {
    const int r = i >> 9, c = i & (DMOD - 1);
    Qs[i] = qp[(size_t)(b * NPTS + row0 + r) * DMOD + c];
  }
  if (tid < 16) { m_i[tid] = -3.0e38f; l_i[tid] = 0.f; }
  v8f acc[4] = {};
  __syncthreads();

  for (int kb = 0; kb < NPTS; kb += 256) {
    // prefetch next chunk (K rows + packed V tiles) while we compute this one
    if (kb + 256 < NPTS) {
      __builtin_prefetch(kp + (size_t)(b * NPTS + kb + 256 + wave * 32) * DMOD, 0, 1);
      __builtin_prefetch(
          vpack + (((size_t)(b * NPTS + kb + 256) >> 5) * 32 + wave * 4) * 512, 0, 1);
    }
    // ---- phase 1: S slab -------------------------------------------------
    {
      const int n0 = wave * 32;
      v8f s0 = {}, s1 = {};
      const u16* kbase = kp + (size_t)(b * NPTS + kb + n0) * DMOD;
#pragma unroll 4
      for (int kk = 0; kk < DMOD; kk += 32) {
        v16bf a  = load_frag_bf16(Qs + kk, DMOD, lane);
        v16bf t0 = load_frag_bf16(kbase + kk, DMOD, lane);             // keys n0..n0+15
        v16bf t1 = load_frag_bf16(kbase + 16 * DMOD + kk, DMOD, lane); // keys +16
        s0 = wmma_bf16(a, t0, s0);
        s1 = wmma_bf16(a, t1, s1);
      }
      const int hf = lane >> 4, r = lane & 15;
#pragma unroll
      for (int j = 0; j < 8; ++j) {
        Ss[(j + hf * 8) * 256 + n0 + r]      = s0[j] * scale;
        Ss[(j + hf * 8) * 256 + n0 + 16 + r] = s1[j] * scale;
      }
    }
    __syncthreads();
    // ---- phase 2: online softmax ----------------------------------------
    {
      const int r = tid >> 4, sub = tid & 15;
      float mx = -3.0e38f;
#pragma unroll
      for (int j = 0; j < 16; ++j) mx = fmaxf(mx, Ss[r * 256 + sub * 16 + j]);
      red[r * 16 + sub] = mx;
    }
    __syncthreads();
    if ((tid & 15) == 0) {
      const int r = tid >> 4;
      float mx = m_i[r];
#pragma unroll
      for (int j = 0; j < 16; ++j) mx = fmaxf(mx, red[r * 16 + j]);
      const float sc = __expf(m_i[r] - mx);
      sc_i[r] = sc;
      l_i[r] *= sc;
      m_i[r] = mx;
    }
    __syncthreads();
    {
      const int r = tid >> 4, sub = tid & 15;
      const float mr = m_i[r];
      float s = 0.f;
#pragma unroll
      for (int j = 0; j < 16; ++j) {
        const int c = sub * 16 + j;
        const float p = __expf(Ss[r * 256 + c] - mr);
        Ps[r * 256 + c] = f2bf(p);
        s += p;
      }
      red[r * 16 + sub] = s;
      // rescale this wave's output accumulators by exp(m_old - m_new) per row
      const int hf = lane >> 4;
#pragma unroll
      for (int j = 0; j < 8; ++j) {
        const float f = sc_i[j + hf * 8];
#pragma unroll
        for (int t = 0; t < 4; ++t) acc[t][j] *= f;
      }
    }
    __syncthreads();
    if ((tid & 15) == 0) {
      const int r = tid >> 4;
      float s = l_i[r];
#pragma unroll
      for (int j = 0; j < 16; ++j) s += red[r * 16 + j];
      l_i[r] = s;
    }
    __syncthreads();
    // ---- phase 3: O += P * V (packed V -> b128 fragment loads) ----------
    {
      const int ct0 = wave * 4; // (wave*64)/16
#pragma unroll 2
      for (int kl = 0; kl < 256; kl += 32) {
        v16bf a = load_frag_bf16(Ps + kl, 256, lane);
        const u16* btile =
            vpack + (((size_t)(b * NPTS + kb + kl) >> 5) * 32 + ct0) * 512;
#pragma unroll
        for (int t = 0; t < 4; ++t) {
          v16bf bb = load_fragB_packed(btile + t * 512, lane);
          acc[t] = wmma_bf16(a, bb, acc[t]);
        }
      }
    }
    __syncthreads();
  }
  // ---- finalize: O / l -> attn output (f32) ------------------------------
  {
    const int c0 = wave * 64, hf = lane >> 4, r = lane & 15;
#pragma unroll
    for (int j = 0; j < 8; ++j) {
      const int row = j + hf * 8;
      const float inv = 1.0f / l_i[row];
#pragma unroll
      for (int t = 0; t < 4; ++t)
        out[(size_t)(b * NPTS + row0 + row) * DMOD + c0 + t * 16 + r] =
            acc[t][j] * inv;
    }
  }
}

// ---------------------------------------------------------------------------
// Host launcher
// ---------------------------------------------------------------------------
extern "C" void kernel_launch(void* const* d_in, const int* in_sizes, int n_in,
                              void* d_out, int out_size, void* d_ws, size_t ws_size,
                              hipStream_t stream) {
  (void)in_sizes; (void)n_in; (void)out_size; (void)ws_size;
  const float* xyz  = (const float*)d_in[0];
  const float* feat = (const float*)d_in[1];
  const float* W1   = (const float*)d_in[2];
  const float* b1   = (const float*)d_in[3];
  const float* W2   = (const float*)d_in[4];
  const float* b2   = (const float*)d_in[5];
  const float* Wd1  = (const float*)d_in[6];
  const float* bd1  = (const float*)d_in[7];
  const float* Wd2  = (const float*)d_in[8];
  const float* bd2  = (const float*)d_in[9];
  const float* Wq   = (const float*)d_in[10];
  const float* Wk   = (const float*)d_in[11];
  const float* Wv   = (const float*)d_in[12];
  // d_in[13] == k (16), hardcoded.

  // Workspace layout (~140 MB total)
  u16* x_bf  = (u16*)d_ws;                         // [NM,512] bf16
  u16* h_bf  = x_bf + (size_t)NM * DMOD;           // [NM,512] bf16
  u16* q_bf  = h_bf + (size_t)NM * DMOD;           // q' = x@Wq + wa (bf16)
  u16* k_bf  = q_bf + (size_t)NM * DMOD;           // k' = x@Wk + wa (bf16)
  u16* v_bf  = k_bf + (size_t)NM * DMOD;           // v' = x@Wv + wa (bf16)
  u16* vpack = v_bf + (size_t)NM * DMOD;           // v' fragment-major (bf16)
  float* wa  = (float*)(vpack + (size_t)NM * DMOD);// weighted_avg f32 [NM,512]
  float* kw  = wa + (size_t)NM * DMOD;             // normalized knn weights [NM,16]
  int* kidx  = (int*)(kw + (size_t)NM * KNN);      // knn indices [NM,16]
  u16* wp1   = (u16*)(kidx + (size_t)NM * KNN);    // packed W1   [64,512]
  u16* wpd2  = wp1  + (size_t)DPTS * DMOD;         // packed Wd2  [512,512]
  u16* wpq   = wpd2 + (size_t)DMOD * DMOD;         // packed Wq
  u16* wpk   = wpq  + (size_t)DMOD * DMOD;         // packed Wk
  u16* wpv   = wpk  + (size_t)DMOD * DMOD;         // packed Wv
  u16* wp2   = wpv  + (size_t)DMOD * DMOD;         // packed W2   [512,64]

  float* res_out  = (float*)d_out;                 // [B,N,64]
  float* attn_out = res_out + (size_t)NM * DPTS;   // [B,N,512]

  // 0) pack all weight matrices to fragment-major bf16
  pack_b_kernel<false><<<(DPTS * DMOD + 255) / 256, 256, 0, stream>>>(W1, wp1, DPTS, DMOD);
  pack_b_kernel<false><<<(DMOD * DMOD + 255) / 256, 256, 0, stream>>>(Wd2, wpd2, DMOD, DMOD);
  pack_b_kernel<false><<<(DMOD * DMOD + 255) / 256, 256, 0, stream>>>(Wq, wpq, DMOD, DMOD);
  pack_b_kernel<false><<<(DMOD * DMOD + 255) / 256, 256, 0, stream>>>(Wk, wpk, DMOD, DMOD);
  pack_b_kernel<false><<<(DMOD * DMOD + 255) / 256, 256, 0, stream>>>(Wv, wpv, DMOD, DMOD);
  pack_b_kernel<false><<<(DMOD * DPTS + 255) / 256, 256, 0, stream>>>(W2, wp2, DMOD, DPTS);

  // 1) kNN + normalized inverse-distance weights
  knn_kernel<<<NM / 8, 256, 0, stream>>>(xyz, kidx, kw);
  // 2) hoisted pos-enc hidden layer (weighted avg before Wd2)
  posenc_kernel<<<NM, 256, 0, stream>>>(xyz, kidx, kw, Wd1, bd1, h_bf);
  // 3) x = feat @ W1 + b1  (f32 A, bf16 out)
  gemm_kernel<false, true, true, false>
      <<<(NM / 16) * (DMOD / 64) / 8, 256, 0, stream>>>(feat, wp1, b1, nullptr,
                                                        x_bf, NM, DMOD, DPTS);
  // 4) wa = h @ Wd2 + bd2  (f32 out)
  gemm_kernel<true, false, true, false>
      <<<(NM / 16) * (DMOD / 64) / 8, 256, 0, stream>>>(h_bf, wpd2, bd2, nullptr,
                                                        wa, NM, DMOD, DMOD);
  // 5) q' / k' / v' = x @ W{q,k,v} + wa  (bf16 out)
  gemm_kernel<true, true, false, true>
      <<<(NM / 16) * (DMOD / 64) / 8, 256, 0, stream>>>(x_bf, wpq, nullptr, wa,
                                                        q_bf, NM, DMOD, DMOD);
  gemm_kernel<true, true, false, true>
      <<<(NM / 16) * (DMOD / 64) / 8, 256, 0, stream>>>(x_bf, wpk, nullptr, wa,
                                                        k_bf, NM, DMOD, DMOD);
  gemm_kernel<true, true, false, true>
      <<<(NM / 16) * (DMOD / 64) / 8, 256, 0, stream>>>(x_bf, wpv, nullptr, wa,
                                                        v_bf, NM, DMOD, DMOD);
  // 5b) pack v' to fragment-major for the P*V phase
  pack_b_kernel<true><<<((int)((size_t)NM * DMOD) + 255) / 256, 256, 0, stream>>>(
      v_bf, vpack, NM, DMOD);
  // 6) flash attention -> attn output (f32, second half of d_out)
  attn_kernel<<<NM / 16, 256, 0, stream>>>(q_bf, k_bf, vpack, attn_out);
  // 7) res = attn @ W2 + b2 + feat  (f32 A from d_out, f32 out)
  gemm_kernel<false, false, true, true>
      <<<(NM / 16) * (DPTS / 64) / 8, 256, 0, stream>>>(attn_out, wp2, b2, feat,
                                                        res_out, NM, DPTS, DMOD);
}
